// MultiHeadAttention_23321672418020
// MI455X (gfx1250) — compile-verified, matrix operand order checked by reading, not run on hardware
//
#include <hip/hip_runtime.h>
#include <hip/hip_bf16.h>
#include <math.h>
#include <stdint.h>

// ---------------------------------------------------------------------------
// Multi-head attention for MI455X (gfx1250, wave32, WMMA).
//   q,k,v:           [B=4, S=2048, 1024] fp32
//   w*_w:            [1024, 1024] fp32 (torch convention, y = x @ W.T + b)
//   out:             [B, S, 1024] fp32
// Pipeline:
//   1) gemm_nt<float,1>  x3 : QKV projections -> f16 [B,H,S,64] (split heads)
//   2) flash-attention       -> f16 ctx [B,S,1024] (heads merged)
//   3) gemm_nt<_Float16,0>   : output projection -> fp32 d_out
// Matrix math: v_wmma_f32_16x16x32_f16. Attention K/Q tiles staged with
// global_load_async_to_lds_b128 (ASYNCcnt); softmax row reductions via DPP16.
// ---------------------------------------------------------------------------

typedef __attribute__((ext_vector_type(16))) _Float16 v16h;
typedef __attribute__((ext_vector_type(8)))  _Float16 v8h;
typedef __attribute__((ext_vector_type(4)))  _Float16 v4h;
typedef __attribute__((ext_vector_type(2)))  _Float16 v2h;
typedef __attribute__((ext_vector_type(8)))  float    v8f;

#define NB   4
#define NS   2048
#define NH   16
#define ND   64
#define DM   1024

__device__ __forceinline__ v8f wmma_f16(v16h a, v16h b, v8f c) {
    // (neg_a, A, neg_b, B, c_mod, C, reuse_a, reuse_b)
    return __builtin_amdgcn_wmma_f32_16x16x32_f16(false, a, false, b, (short)0, c,
                                                  false, false);
}

// ---- 16-lane row reductions (DPP16: pure VALU, no LDS round trips) --------
#if __has_builtin(__builtin_amdgcn_mov_dpp)
template <int CTRL>
__device__ __forceinline__ float dpp_movf(float x) {
    const int y = __builtin_amdgcn_mov_dpp(__builtin_bit_cast(int, x), CTRL,
                                           0xF, 0xF, true);
    return __builtin_bit_cast(float, y);
}
__device__ __forceinline__ float row16_max(float x) {
    x = fmaxf(x, dpp_movf<0x0B1>(x));   // quad_perm [1,0,3,2]  (xor 1)
    x = fmaxf(x, dpp_movf<0x04E>(x));   // quad_perm [2,3,0,1]  (xor 2)
    x = fmaxf(x, dpp_movf<0x141>(x));   // row_half_mirror      (xor 7)
    x = fmaxf(x, dpp_movf<0x140>(x));   // row_mirror           (xor 15)
    return x;
}
__device__ __forceinline__ float row16_sum(float x) {
    x += dpp_movf<0x0B1>(x);
    x += dpp_movf<0x04E>(x);
    x += dpp_movf<0x141>(x);
    x += dpp_movf<0x140>(x);
    return x;
}
#else
__device__ __forceinline__ float row16_max(float x) {
    for (int m = 8; m >= 1; m >>= 1) x = fmaxf(x, __shfl_xor(x, m, 32));
    return x;
}
__device__ __forceinline__ float row16_sum(float x) {
    for (int m = 8; m >= 1; m >>= 1) x += __shfl_xor(x, m, 32);
    return x;
}
#endif

// ---- gfx1250 async copy: global memory -> LDS, 16B per lane ---------------
__device__ __forceinline__ void async_b128(void* lds_dst, const void* gsrc) {
    const uint32_t laddr = (uint32_t)(uintptr_t)lds_dst;     // low 32 = LDS off
    const uint64_t gaddr = (uint64_t)(uintptr_t)gsrc;
    asm volatile("global_load_async_to_lds_b128 %0, %1, off"
                 :
                 : "v"(laddr), "v"(gaddr)
                 : "memory");
}
__device__ __forceinline__ void async_wait0() {
    asm volatile("s_wait_asynccnt 0x0" ::: "memory");
}

// Load a 16x32 (rows x k) f16 fragment from an LDS tile into WMMA A/B layout.
// lanes 0-15 -> k base 0, lanes 16-31 -> k base 8; elems 0..7 = k0..k0+7,
// elems 8..15 = 16+k0 .. 16+k0+7. ld (halves) must be a multiple of 8.
__device__ __forceinline__ v16h lds_frag(const _Float16* base, int row0, int col0,
                                         int ld) {
    const int lane = threadIdx.x & 31;
    const int r    = lane & 15;
    const int kh   = (lane & 16) >> 1;      // 0 or 8
    const _Float16* p = base + (row0 + r) * ld + col0 + kh;
    v8h lo = *(const v8h*)(p);
    v8h hi = *(const v8h*)(p + 16);
    v16h out;
#pragma unroll
    for (int i = 0; i < 8; ++i) { out[i] = lo[i]; out[i + 8] = hi[i]; }
    return out;
}

// 4 consecutive source elements -> 4 packed halves.
__device__ __forceinline__ v4h cvt4(const float* p) {
    const float4 v = *(const float4*)p;
    v4h h;
    h[0] = (_Float16)v.x; h[1] = (_Float16)v.y;
    h[2] = (_Float16)v.z; h[3] = (_Float16)v.w;
    return h;
}
__device__ __forceinline__ v4h cvt4(const _Float16* p) {
    return *(const v4h*)p;
}

// Cooperatively (256 threads) load a 128x32 tile from row-major src (leading
// dim ld_src) into LDS (leading dim 40 halves), converting to f16.
template <typename AT>
__device__ __forceinline__ void load_tile(_Float16* dst, const AT* src, int ld_src) {
    const int tid = threadIdx.x;
#pragma unroll
    for (int i = 0; i < 4; ++i) {
        const int c   = tid + i * 256;   // 0..1023 chunks of 4 elems
        const int row = c >> 3;
        const int c4  = (c & 7) * 4;
        *(v4h*)(dst + row * 40 + c4) = cvt4(src + (size_t)row * ld_src + c4);
    }
}

// ---------------------------------------------------------------------------
// NT GEMM: C[m,n] = sum_k A[m,k] * W[n,k] + bias[n]
//   A: [M,K] row-major (fp32 or f16), W: [N,K] row-major fp32.
// EPI = 0: fp32 output, row-major [M,N]
// EPI = 1: f16 output in split-head layout [B, H, S, 64]   (m=b*S+s, n=h*64+d)
// Block: 256 threads (8 waves), tile 128x128x32, double-buffered LDS.
// ---------------------------------------------------------------------------
template <typename AT, int EPI>
__global__ __launch_bounds__(256) void gemm_nt_kernel(
    const AT* __restrict__ A, const float* __restrict__ W,
    const float* __restrict__ bias, void* __restrict__ outp, int M, int N, int K) {
    __shared__ _Float16 As[2][128 * 40];
    __shared__ _Float16 Bs[2][128 * 40];

    const int n0 = blockIdx.x * 128;
    const int m0 = blockIdx.y * 128;
    const int w  = threadIdx.x >> 5;
    const int wm = (w >> 2) * 64;   // wave row offset: 0 / 64
    const int wn = (w & 3) * 32;    // wave col offset: 0 / 32 / 64 / 96

    v8f acc[4][2];
#pragma unroll
    for (int i = 0; i < 4; ++i)
#pragma unroll
        for (int j = 0; j < 2; ++j) acc[i][j] = (v8f)(0.0f);

    const AT*    Atile = A + (size_t)m0 * K;
    const float* Wtile = W + (size_t)n0 * K;
    const int    nk    = K >> 5;

    load_tile<AT>(As[0], Atile, K);
    load_tile<float>(Bs[0], Wtile, K);

    for (int kt = 0; kt < nk; ++kt) {
        __syncthreads();
        const int cur = kt & 1;
        if (kt + 1 < nk) {
            load_tile<AT>(As[cur ^ 1], Atile + (kt + 1) * 32, K);
            load_tile<float>(Bs[cur ^ 1], Wtile + (kt + 1) * 32, K);
        }
        if (kt + 2 < nk) {   // gfx1250 global_prefetch_b8 path
            __builtin_prefetch(Atile + (size_t)(threadIdx.x >> 1) * K + (kt + 2) * 32, 0, 1);
            __builtin_prefetch(Wtile + (size_t)(threadIdx.x >> 1) * K + (kt + 2) * 32, 0, 1);
        }

        v16h af[4], bf[2];
#pragma unroll
        for (int i = 0; i < 4; ++i) af[i] = lds_frag(As[cur], wm + i * 16, 0, 40);
#pragma unroll
        for (int j = 0; j < 2; ++j) bf[j] = lds_frag(Bs[cur], wn + j * 16, 0, 40);
#pragma unroll
        for (int i = 0; i < 4; ++i)
#pragma unroll
            for (int j = 0; j < 2; ++j) acc[i][j] = wmma_f16(af[i], bf[j], acc[i][j]);
    }

    // Epilogue. C layout: lane -> N (lane&15), VGPR r -> M = r + 8*(lane>=16).
    const int lane = threadIdx.x & 31;
    const int ln   = lane & 15;
    const int mhi  = (lane >> 4) << 3;
#pragma unroll
    for (int i = 0; i < 4; ++i) {
#pragma unroll
        for (int j = 0; j < 2; ++j) {
            const int   ng = n0 + wn + j * 16 + ln;
            const float bv = bias[ng];
#pragma unroll
            for (int r = 0; r < 8; ++r) {
                const int   mg  = m0 + wm + i * 16 + mhi + r;
                const float val = acc[i][j][r] + bv;
                if (EPI == 0) {
                    ((float*)outp)[(size_t)mg * N + ng] = val;
                } else {
                    const int b = mg >> 11, s = mg & (NS - 1);
                    const int h = ng >> 6,  d = ng & (ND - 1);
                    ((_Float16*)outp)[((((size_t)b * NH + h) * NS) + s) * ND + d] =
                        (_Float16)val;
                }
            }
        }
    }
}

// ---------------------------------------------------------------------------
// Flash attention over f16 [B,H,S,64] Q/K/V. One block = 128 query rows of one
// (b,h); 8 waves x 16 rows. Streams keys/values in chunks of 64 with online
// softmax. Q/K tiles staged via async global->LDS; V transposed via VGPRs with
// packed b32 LDS stores. Output: f16 ctx[B,S,1024] with heads merged.
// ---------------------------------------------------------------------------
__global__ __launch_bounds__(256) void attention_kernel(
    const _Float16* __restrict__ Qp, const _Float16* __restrict__ Kp,
    const _Float16* __restrict__ Vp, _Float16* __restrict__ ctx) {
    __shared__ _Float16 Qs[128 * 72];
    __shared__ _Float16 Ks[64 * 72];
    __shared__ _Float16 Vt[64 * 72];          // transposed: Vt[d][kk]
    __shared__ _Float16 Ps[8 * 16 * 72];      // per-wave P staging (C -> A layout)

    const int bh   = blockIdx.y;              // 0..63
    const int b    = bh >> 4, h = bh & 15;
    const int q0   = blockIdx.x * 128;
    const int tid  = threadIdx.x;
    const int w    = tid >> 5;
    const int lane = tid & 31;
    const size_t base = (size_t)bh * NS * ND;

    // Async-load 128x64 Q tile (f16, already split-head layout).
#pragma unroll
    for (int i = 0; i < 4; ++i) {
        const int c = tid + i * 256;          // 0..1023 chunks of 8 halves
        const int row = c >> 3, c8 = (c & 7) * 8;
        async_b128(Qs + row * 72 + c8, Qp + base + (size_t)(q0 + row) * ND + c8);
    }
    async_wait0();
    __syncthreads();

    v16h qf[2];
    qf[0] = lds_frag(Qs, w * 16, 0, 72);
    qf[1] = lds_frag(Qs, w * 16, 32, 72);

    v8f oacc[4];
#pragma unroll
    for (int t = 0; t < 4; ++t) oacc[t] = (v8f)(0.0f);
    float mrow[8], lrow[8];
#pragma unroll
    for (int r = 0; r < 8; ++r) { mrow[r] = -INFINITY; lrow[r] = 0.0f; }

    for (int kt = 0; kt < NS / 64; ++kt) {
        __syncthreads();   // previous iteration done reading Ks/Vt
        // K chunk 64x64 -> Ks row-major (async, overlaps V transpose below).
#pragma unroll
        for (int i = 0; i < 2; ++i) {
            const int c = tid + i * 256;      // 0..511 chunks of 8 halves
            const int row = c >> 3, c8 = (c & 7) * 8;
            async_b128(Ks + row * 72 + c8,
                       Kp + base + (size_t)(kt * 64 + row) * ND + c8);
        }
        // V chunk 64x64 -> Vt transposed. Each thread handles 2 consecutive
        // key rows so the transposed writes pack into 32-bit DS stores.
        {
            const int rp = tid >> 3;              // 0..31 -> rows 2rp, 2rp+1
            const int c8 = (tid & 7) * 8;
            const _Float16* src =
                Vp + base + (size_t)(kt * 64 + rp * 2) * ND + c8;
            const v8h va = *(const v8h*)(src);
            const v8h vb = *(const v8h*)(src + ND);
#pragma unroll
            for (int e = 0; e < 8; ++e) {
                v2h pk; pk[0] = va[e]; pk[1] = vb[e];
                *(v2h*)(Vt + (c8 + e) * 72 + rp * 2) = pk;
            }
        }
        async_wait0();
        __syncthreads();

        // S = Q * K^T for this wave's 16 rows x 64 keys.
        v8f sacc[4];
#pragma unroll
        for (int j = 0; j < 4; ++j) sacc[j] = (v8f)(0.0f);
#pragma unroll
        for (int c = 0; c < 2; ++c)
#pragma unroll
            for (int j = 0; j < 4; ++j) {
                const v16h kf = lds_frag(Ks, j * 16, c * 32, 72);
                sacc[j] = wmma_f16(qf[c], kf, sacc[j]);
            }

        // Online softmax per row slot r (row m = r + 8*(lane>=16)).
#pragma unroll
        for (int r = 0; r < 8; ++r) {
            float m4 = -INFINITY;
#pragma unroll
            for (int j = 0; j < 4; ++j) {
                sacc[j][r] *= 0.125f;               // 1/sqrt(64)
                m4 = fmaxf(m4, sacc[j][r]);
            }
            m4 = row16_max(m4);
            const float nm   = fmaxf(mrow[r], m4);
            const float corr = __expf(mrow[r] - nm);
            mrow[r] = nm;
            lrow[r] *= corr;
#pragma unroll
            for (int t = 0; t < 4; ++t) oacc[t][r] *= corr;
            float psum = 0.0f;
#pragma unroll
            for (int j = 0; j < 4; ++j) {
                const float p = __expf(sacc[j][r] - nm);
                sacc[j][r] = p;
                psum += p;
            }
            lrow[r] += row16_sum(psum);
        }

        // Stage P (C layout) into LDS row-major so it can re-enter as A frags.
        _Float16* pw = Ps + w * 16 * 72;
        const int mhi = (lane >> 4) << 3;
#pragma unroll
        for (int j = 0; j < 4; ++j)
#pragma unroll
            for (int r = 0; r < 8; ++r)
                pw[(mhi + r) * 72 + j * 16 + (lane & 15)] = (_Float16)sacc[j][r];

        const v16h pf0 = lds_frag(pw, 0, 0, 72);
        const v16h pf1 = lds_frag(pw, 0, 32, 72);
#pragma unroll
        for (int t = 0; t < 4; ++t) {
            v16h vf = lds_frag(Vt, t * 16, 0, 72);
            oacc[t] = wmma_f16(pf0, vf, oacc[t]);
            vf = lds_frag(Vt, t * 16, 32, 72);
            oacc[t] = wmma_f16(pf1, vf, oacc[t]);
        }
    }

    // Normalize and write ctx (heads merged): ctx[(b*S+s)*1024 + h*64 + d].
    const int mhi = (lane >> 4) << 3;
#pragma unroll
    for (int t = 0; t < 4; ++t)
#pragma unroll
        for (int r = 0; r < 8; ++r) {
            const int   m   = w * 16 + mhi + r;
            const float val = oacc[t][r] / lrow[r];
            const int   sg  = q0 + m;
            ctx[((size_t)(b * NS + sg)) * DM + h * ND + t * 16 + (lane & 15)] =
                (_Float16)val;
        }
}

// ---------------------------------------------------------------------------
extern "C" void kernel_launch(void* const* d_in, const int* in_sizes, int n_in,
                              void* d_out, int out_size, void* d_ws, size_t ws_size,
                              hipStream_t stream) {
    const float* q      = (const float*)d_in[0];
    const float* k      = (const float*)d_in[1];
    const float* v      = (const float*)d_in[2];
    const float* wq_w   = (const float*)d_in[3];
    const float* wq_b   = (const float*)d_in[4];
    const float* wk_w   = (const float*)d_in[5];
    const float* wk_b   = (const float*)d_in[6];
    const float* wv_w   = (const float*)d_in[7];
    const float* wv_b   = (const float*)d_in[8];
    const float* proj_w = (const float*)d_in[9];
    const float* proj_b = (const float*)d_in[10];

    const size_t elems = (size_t)NB * NH * NS * ND;   // 8.39M f16 per tensor
    _Float16* Qp  = (_Float16*)d_ws;
    _Float16* Kp  = Qp + elems;
    _Float16* Vp  = Kp + elems;
    _Float16* ctx = Vp + elems;                       // [B,S,1024] f16

    const int M = NB * NS;   // 8192
    const dim3 blk(256);
    const dim3 grid_g(DM / 128, M / 128);             // (8, 64)

    gemm_nt_kernel<float, 1><<<grid_g, blk, 0, stream>>>(q, wq_w, wq_b, Qp, M, DM, DM);
    gemm_nt_kernel<float, 1><<<grid_g, blk, 0, stream>>>(k, wk_w, wk_b, Kp, M, DM, DM);
    gemm_nt_kernel<float, 1><<<grid_g, blk, 0, stream>>>(v, wv_w, wv_b, Vp, M, DM, DM);

    attention_kernel<<<dim3(NS / 128, NB * NH), blk, 0, stream>>>(Qp, Kp, Vp, ctx);

    gemm_nt_kernel<_Float16, 0><<<grid_g, blk, 0, stream>>>(ctx, proj_w, proj_b,
                                                            d_out, M, DM, DM);
}
